// PVMLayer_shifted_70875550319174
// MI455X (gfx1250) — compile-verified, matrix operand order checked by reading, not run on hardware
//
#include <hip/hip_runtime.h>
#include <hip/hip_bf16.h>

// ---------------- problem constants ----------------
#define B_IMG   4
#define C_CH    256
#define L_SEQ   4096          // 64*64
#define NBATCH  16            // 4 branches * B
#define DMODEL  64
#define DINNER  128
#define DSTATE  16
#define DTRANK  4
#define M1      (NBATCH * L_SEQ)   // 65536 mamba token rows
#define MBL     (B_IMG * L_SEQ)    // 16384 image token rows
#define DBLP    48            // dt_rank + 2*d_state (36) padded to 3 WMMA tiles
#define CHUNK   64
#define NCHUNK  (L_SEQ / CHUNK)    // 64

typedef __attribute__((ext_vector_type(16))) _Float16 v16h;
typedef __attribute__((ext_vector_type(8)))  _Float16 v8h;
typedef __attribute__((ext_vector_type(8)))  float    v8f;

// ---------------------------------------------------------------------------
// WMMA fragment load, 16-bit A-matrix 16x32 layout (ISA 7.12.2):
//   lane L holds row M = L&15; half = L>>4 selects K-block.
//   v16h elems 0..7  -> K = kbase + 8*half + 0..7      (contiguous 16B)
//   v16h elems 8..15 -> K = kbase + 16 + 8*half + 0..7 (contiguous 16B)
// B operand (32x16 K-major) is loaded from a row-major [N,K] weight matrix
// with the mirrored swizzle: lane holds column N = L&15 of B == row N of W.
// Works for both global and LDS-resident source (compiler picks ds_load).
// ---------------------------------------------------------------------------
__device__ __forceinline__ v16h load_frag16(const _Float16* base,
                                            int ld, int rowBase, int kbase, int lane) {
    const _Float16* p = base + (size_t)(rowBase + (lane & 15)) * ld
                             + kbase + ((lane >> 4) << 3);
    v8h lo = *(const v8h*)(p);
    v8h hi = *(const v8h*)(p + 16);
    v16h r;
#pragma unroll
    for (int i = 0; i < 8; ++i) { r[i] = lo[i]; r[i + 8] = hi[i]; }
    return r;
}

// ---------------------------------------------------------------------------
// Generic WMMA GEMM:  C[M,N] = A[M,K] * W[N,K]^T   (f16 in, f32 accum)
// One wave -> 16 x (NT*16) output tile. Block = 256 thr = 8 waves stacked in M.
// Weight tile (NT*16 rows x K) is staged once per block into LDS with
// GLOBAL_LOAD_ASYNC_TO_LDS_B128 (ASYNCcnt) and consumed via ds_load_b128.
// All NT B-fragments are loaded before the NT WMMAs so the LDS loads cluster
// behind a single wait instead of a WAR-serialized load/wmma chain.
// EPI: 0 = plain store | 1 = + skip_scale * skip[row,col]
//      2 = + bias[col], scatter-store NCHW-transposed into d_out
// LDC is a compile-time constant so epilogue addresses fold to imm offsets.
// ---------------------------------------------------------------------------
template <int NT, int KT, int EPI, int LDC>
__global__ void __launch_bounds__(256)
wmma_gemm_kernel(const _Float16* __restrict__ A, const _Float16* __restrict__ W,
                 float* __restrict__ C,
                 const float* __restrict__ e0, const float* __restrict__ e1) {
    __shared__ __align__(16) _Float16 wlds[NT * 16 * KT];

    const int lane  = threadIdx.x & 31;
    const int wave  = threadIdx.x >> 5;
    const int mBase = (blockIdx.x * 8 + wave) * 16;
    const int nBase = blockIdx.y * (NT * 16);

    // ---- async-stage the (contiguous) weight tile into LDS ----
    {
        const char* wsrc = (const char*)(W + (size_t)nBase * KT);
        // low 32 bits of a generic pointer to __shared__ == wave-relative LDS offset
        const unsigned ldsBase = (unsigned)(uintptr_t)&wlds[0];
#pragma unroll
        for (int ofs = 0; ofs < NT * 16 * KT * 2; ofs += 256 * 16) {
            const unsigned laddr = ldsBase + ofs + threadIdx.x * 16;
            const char*    gaddr = wsrc + ofs + threadIdx.x * 16;
            asm volatile("global_load_async_to_lds_b128 %0, %1, off"
                         :: "v"(laddr), "v"(gaddr) : "memory");
        }
        asm volatile("s_wait_asynccnt 0x0" ::: "memory");
        __syncthreads();
    }

    v8f acc[NT];
#pragma unroll
    for (int t = 0; t < NT; ++t) acc[t] = (v8f)0.0f;

#pragma unroll
    for (int k0 = 0; k0 < KT; k0 += 32) {
        v16h a = load_frag16(A, KT, mBase, k0, lane);
        v16h bf[NT];
#pragma unroll
        for (int t = 0; t < NT; ++t) bf[t] = load_frag16(wlds, KT, t * 16, k0, lane);
#pragma unroll
        for (int t = 0; t < NT; ++t)
            acc[t] = __builtin_amdgcn_wmma_f32_16x16x32_f16(
                         false, a, false, bf[t], (short)0, acc[t], false, false);
    }

    // C/D layout: VGPR r -> M = r + 8*(lane>>4), N = lane&15  (ISA 7.12.2)
    const int rowB = mBase + ((lane >> 4) << 3);    // first row this lane owns
#pragma unroll
    for (int t = 0; t < NT; ++t) {
        const int col = nBase + t * 16 + (lane & 15);
        if (EPI == 0) {
            float* cp = C + (size_t)rowB * LDC + col;
#pragma unroll
            for (int r = 0; r < 8; ++r) cp[r * LDC] = acc[t][r];
        } else if (EPI == 1) {
            float* cp = C + (size_t)rowB * LDC + col;
            const float* sp = e1 + (size_t)rowB * LDC + col;
            const float ss = e0[0];
#pragma unroll
            for (int r = 0; r < 8; ++r) cp[r * LDC] = acc[t][r] + ss * sp[r * LDC];
        } else {
            const int b = rowB >> 12;               // tile never crosses an L boundary
            const int l = rowB & (L_SEQ - 1);
            float* cp = C + ((size_t)b * C_CH + col) * L_SEQ + l;
            const float bias = e0[col];
#pragma unroll
            for (int r = 0; r < 8; ++r) cp[r] = acc[t][r] + bias;  // stride-1 in l
        }
    }
}

// ---------------- f32 -> f16 packs ----------------
__global__ void cvt_f16_kernel(const float* __restrict__ in, _Float16* __restrict__ out, int n) {
    int i = blockIdx.x * 256 + threadIdx.x;
    if (i < n) out[i] = (_Float16)in[i];
}
__global__ void pad_xproj_kernel(const float* __restrict__ w, _Float16* __restrict__ out) {
    int i = blockIdx.x * 256 + threadIdx.x;       // 48*128
    if (i >= DBLP * DINNER) return;
    int r = i / DINNER, c = i % DINNER;
    out[i] = (r < (DTRANK + 2 * DSTATE)) ? (_Float16)w[r * DINNER + c] : (_Float16)0.0f;
}

// ---------------- LayerNorm + channel shuffle -> stacked [16,L,64] ----------
__device__ __forceinline__ void shuffle_map(int c, int& r, int& j) {
    if      (c < 32)  { r = 0; j = c + 32;  }   // p0 -> tail of b1
    else if (c < 96)  { r = 1; j = c - 32;  }   // p1,p2
    else if (c < 160) { r = 2; j = c - 96;  }   // p3,p4
    else if (c < 224) { r = 3; j = c - 160; }   // p5,p6
    else              { r = 0; j = c - 224; }   // p7 -> head of b1
}

__global__ void __launch_bounds__(256)
ln_shuffle_kernel(const float* __restrict__ x, const float* __restrict__ g,
                  const float* __restrict__ bt, float* __restrict__ stacked_f,
                  _Float16* __restrict__ stacked_h) {
    const int bl = blockIdx.x;               // B*L
    const int b  = bl >> 12, l = bl & (L_SEQ - 1);
    const int c  = threadIdx.x;
    __shared__ float red[256];

    float v = x[((size_t)b * C_CH + c) * L_SEQ + l];
    red[c] = v; __syncthreads();
    for (int s = 128; s > 0; s >>= 1) { if (c < s) red[c] += red[c + s]; __syncthreads(); }
    const float mu = red[0] * (1.0f / 256.0f); __syncthreads();
    const float dv = v - mu;
    red[c] = dv * dv; __syncthreads();
    for (int s = 128; s > 0; s >>= 1) { if (c < s) red[c] += red[c + s]; __syncthreads(); }
    const float rs = rsqrtf(red[0] * (1.0f / 256.0f) + 1e-5f);
    const float xn = dv * rs * g[c] + bt[c];

    int r, j; shuffle_map(c, r, j);
    const size_t idx = (((size_t)(r * 4 + b)) * L_SEQ + l) * DMODEL + j;
    stacked_f[idx] = xn;
    stacked_h[idx] = (_Float16)xn;
}

// ---------------- causal depthwise conv1d + SiLU ----------------
__global__ void __launch_bounds__(256)
conv_silu_kernel(const float* __restrict__ xz, const float* __restrict__ cw,
                 const float* __restrict__ cb, float* __restrict__ u_f,
                 _Float16* __restrict__ u_h) {
    const size_t i = (size_t)blockIdx.x * 256 + threadIdx.x;   // M1*DINNER
    const int d = (int)(i & (DINNER - 1));
    const size_t bl = i >> 7;
    const int l = (int)(bl & (L_SEQ - 1));
    const size_t bb = bl >> 12;
    float a = cb[d];
#pragma unroll
    for (int k = 0; k < 4; ++k) {
        const int lt = l - 3 + k;
        if (lt >= 0) a += cw[d * 4 + k] * xz[((size_t)bb * L_SEQ + lt) * 256 + d];
    }
    const float s = a / (1.0f + __expf(-a));
    u_f[i] = s;
    u_h[i] = (_Float16)s;
}

// ---------------- delta = softplus(dt @ dt_w^T + dt_b) ----------------
__global__ void __launch_bounds__(256)
delta_kernel(const float* __restrict__ dbl, const float* __restrict__ dtw,
             const float* __restrict__ dtb, float* __restrict__ delta) {
    const size_t i = (size_t)blockIdx.x * 256 + threadIdx.x;   // M1*DINNER
    const int d = (int)(i & (DINNER - 1));
    const size_t row = i >> 7;
    const float* dt = dbl + row * DBLP;
    float a = dtb[d];
#pragma unroll
    for (int k = 0; k < 4; ++k) a += dtw[d * 4 + k] * dt[k];
    delta[i] = (a > 20.0f) ? a : log1pf(__expf(a));
}

// ---------------- selective scan: 3-phase chunked linear recurrence --------
__global__ void __launch_bounds__(128)
scan_phase1(const float* __restrict__ dbl, const float* __restrict__ delta,
            const float* __restrict__ u_f, const float* __restrict__ A_log,
            float* __restrict__ Pc, float* __restrict__ Hc) {
    const int bb = blockIdx.x >> 6, ch = blockIdx.x & 63;
    const int d  = threadIdx.x;
    __shared__ float Bsh[CHUNK * DSTATE];
    const size_t row0 = (size_t)bb * L_SEQ + ch * CHUNK;
    for (int idx = threadIdx.x; idx < CHUNK * DSTATE; idx += 128) {
        const int t = idx >> 4, n = idx & 15;
        Bsh[idx] = dbl[(row0 + t) * DBLP + DTRANK + n];
    }
    __syncthreads();

    float An[DSTATE], h[DSTATE], P[DSTATE];
#pragma unroll
    for (int n = 0; n < DSTATE; ++n) {
        An[n] = -__expf(A_log[d * DSTATE + n]); h[n] = 0.0f; P[n] = 1.0f;
    }
    for (int t = 0; t < CHUNK; ++t) {
        const size_t e = (row0 + t) * DINNER + d;
        const float dl = delta[e];
        const float du = dl * u_f[e];
#pragma unroll
        for (int n = 0; n < DSTATE; ++n) {
            const float w = __expf(dl * An[n]);
            h[n] = w * h[n] + du * Bsh[t * DSTATE + n];
            P[n] *= w;
        }
    }
    const size_t o = ((((size_t)bb * NCHUNK + ch) * DINNER) + d) * DSTATE;
#pragma unroll
    for (int n = 0; n < DSTATE; ++n) { Pc[o + n] = P[n]; Hc[o + n] = h[n]; }
}

__global__ void __launch_bounds__(256)
scan_phase2(const float* __restrict__ Pc, const float* __restrict__ Hc,
            float* __restrict__ Hstart) {
    const int i = blockIdx.x * 256 + threadIdx.x;   // NBATCH*DINNER*DSTATE = 32768
    const int n = i & 15;
    const int dn = i >> 4;
    const int d = dn & (DINNER - 1);
    const int bb = dn >> 7;
    float h = 0.0f;
    for (int c = 0; c < NCHUNK; ++c) {
        const size_t o = ((((size_t)bb * NCHUNK + c) * DINNER) + d) * DSTATE + n;
        Hstart[o] = h;
        h = Pc[o] * h + Hc[o];
    }
}

__global__ void __launch_bounds__(128)
scan_phase3(const float* __restrict__ dbl, const float* __restrict__ delta,
            const float* __restrict__ u_f, const float* __restrict__ xz,
            const float* __restrict__ A_log, const float* __restrict__ Dp,
            const float* __restrict__ Hstart, _Float16* __restrict__ ycore_h) {
    const int bb = blockIdx.x >> 6, ch = blockIdx.x & 63;
    const int d  = threadIdx.x;
    __shared__ float Bsh[CHUNK * DSTATE];
    __shared__ float Csh[CHUNK * DSTATE];
    const size_t row0 = (size_t)bb * L_SEQ + ch * CHUNK;
    for (int idx = threadIdx.x; idx < CHUNK * DSTATE; idx += 128) {
        const int t = idx >> 4, n = idx & 15;
        Bsh[idx] = dbl[(row0 + t) * DBLP + DTRANK + n];
        Csh[idx] = dbl[(row0 + t) * DBLP + DTRANK + DSTATE + n];
    }
    __syncthreads();

    float An[DSTATE], h[DSTATE];
    const size_t o = ((((size_t)bb * NCHUNK + ch) * DINNER) + d) * DSTATE;
#pragma unroll
    for (int n = 0; n < DSTATE; ++n) {
        An[n] = -__expf(A_log[d * DSTATE + n]);
        h[n]  = Hstart[o + n];
    }
    const float Dd = Dp[d];
    for (int t = 0; t < CHUNK; ++t) {
        const size_t row = row0 + t;
        const size_t e = row * DINNER + d;
        const float dl = delta[e];
        const float uu = u_f[e];
        const float du = dl * uu;
        float y = 0.0f;
#pragma unroll
        for (int n = 0; n < DSTATE; ++n) {
            const float w = __expf(dl * An[n]);
            h[n] = w * h[n] + du * Bsh[t * DSTATE + n];
            y += h[n] * Csh[t * DSTATE + n];
        }
        const float z = xz[row * 256 + DINNER + d];
        const float sz = z / (1.0f + __expf(-z));
        ycore_h[e] = (_Float16)((y + uu * Dd) * sz);
    }
}

// ---------------- regroup branches + LayerNorm -> f16 [B*L, 256] ----------
__global__ void __launch_bounds__(256)
regroup_ln_kernel(const float* __restrict__ ym, const float* __restrict__ g,
                  const float* __restrict__ bt, _Float16* __restrict__ xm_h) {
    const int bl = blockIdx.x;
    const int b  = bl >> 12, l = bl & (L_SEQ - 1);
    const int c  = threadIdx.x;
    __shared__ float red[256];

    int r, j; shuffle_map(c, r, j);   // inverse shuffle == forward map
    float v = ym[(((size_t)(r * 4 + b)) * L_SEQ + l) * DMODEL + j];
    red[c] = v; __syncthreads();
    for (int s = 128; s > 0; s >>= 1) { if (c < s) red[c] += red[c + s]; __syncthreads(); }
    const float mu = red[0] * (1.0f / 256.0f); __syncthreads();
    const float dv = v - mu;
    red[c] = dv * dv; __syncthreads();
    for (int s = 128; s > 0; s >>= 1) { if (c < s) red[c] += red[c + s]; __syncthreads(); }
    const float rs = rsqrtf(red[0] * (1.0f / 256.0f) + 1e-5f);
    xm_h[(size_t)bl * C_CH + c] = (_Float16)(dv * rs * g[c] + bt[c]);
}

// ===========================================================================
extern "C" void kernel_launch(void* const* d_in, const int* in_sizes, int n_in,
                              void* d_out, int out_size, void* d_ws, size_t ws_size,
                              hipStream_t stream) {
    (void)in_sizes; (void)n_in; (void)out_size; (void)ws_size;

    const float* x        = (const float*)d_in[0];
    const float* norm_g   = (const float*)d_in[1];
    const float* norm_b   = (const float*)d_in[2];
    const float* skip_s   = (const float*)d_in[3];
    const float* proj_w   = (const float*)d_in[4];
    const float* proj_b   = (const float*)d_in[5];
    const float* in_projw = (const float*)d_in[6];
    const float* conv_w   = (const float*)d_in[7];
    const float* conv_b   = (const float*)d_in[8];
    const float* x_projw  = (const float*)d_in[9];
    const float* dt_w     = (const float*)d_in[10];
    const float* dt_b     = (const float*)d_in[11];
    const float* A_log    = (const float*)d_in[12];
    const float* Dp       = (const float*)d_in[13];
    const float* out_w    = (const float*)d_in[14];
    float* out = (float*)d_out;

    // ---- workspace sub-allocation (256B aligned) ----
    char* ws = (char*)d_ws;
    size_t off = 0;
    auto alloc = [&](size_t bytes) -> void* {
        off = (off + 255) & ~(size_t)255;
        void* p = ws + off;
        off += bytes;
        return p;
    };
    float*    stacked_f = (float*)   alloc((size_t)M1 * DMODEL * 4);
    _Float16* stacked_h = (_Float16*)alloc((size_t)M1 * DMODEL * 2);
    float*    xz        = (float*)   alloc((size_t)M1 * 256 * 4);
    float*    u_f       = (float*)   alloc((size_t)M1 * DINNER * 4);
    _Float16* u_h       = (_Float16*)alloc((size_t)M1 * DINNER * 2);
    float*    dbl       = (float*)   alloc((size_t)M1 * DBLP * 4);
    float*    delta     = (float*)   alloc((size_t)M1 * DINNER * 4);
    const size_t carry  = (size_t)NBATCH * NCHUNK * DINNER * DSTATE;
    float*    Pc        = (float*)   alloc(carry * 4);
    float*    Hc        = (float*)   alloc(carry * 4);
    float*    Hstart    = (float*)   alloc(carry * 4);
    _Float16* ycore_h   = (_Float16*)alloc((size_t)M1 * DINNER * 2);
    float*    ym        = (float*)   alloc((size_t)M1 * DMODEL * 4);
    _Float16* xm_h      = (_Float16*)alloc((size_t)MBL * C_CH * 2);
    _Float16* ipw_h     = (_Float16*)alloc((size_t)256 * DMODEL * 2);
    _Float16* xpw_h     = (_Float16*)alloc((size_t)DBLP * DINNER * 2);
    _Float16* ow_h      = (_Float16*)alloc((size_t)DMODEL * DINNER * 2);
    _Float16* pw_h      = (_Float16*)alloc((size_t)C_CH * C_CH * 2);

    // ---- weight packs ----
    cvt_f16_kernel<<<(256 * DMODEL + 255) / 256, 256, 0, stream>>>(in_projw, ipw_h, 256 * DMODEL);
    pad_xproj_kernel<<<(DBLP * DINNER + 255) / 256, 256, 0, stream>>>(x_projw, xpw_h);
    cvt_f16_kernel<<<(DMODEL * DINNER + 255) / 256, 256, 0, stream>>>(out_w, ow_h, DMODEL * DINNER);
    cvt_f16_kernel<<<(C_CH * C_CH + 255) / 256, 256, 0, stream>>>(proj_w, pw_h, C_CH * C_CH);

    // ---- 1) LayerNorm + branch shuffle ----
    ln_shuffle_kernel<<<MBL, 256, 0, stream>>>(x, norm_g, norm_b, stacked_f, stacked_h);

    // ---- 2) in_proj GEMM: [65536,64] x [256,64]^T -> xz [65536,256] ----
    wmma_gemm_kernel<4, DMODEL, 0, 256><<<dim3(M1 / 128, 256 / 64), 256, 0, stream>>>(
        stacked_h, ipw_h, xz, nullptr, nullptr);

    // ---- 3) causal depthwise conv + SiLU -> u ----
    conv_silu_kernel<<<(size_t)M1 * DINNER / 256, 256, 0, stream>>>(xz, conv_w, conv_b, u_f, u_h);

    // ---- 4) x_proj GEMM: [65536,128] x [48,128]^T -> dbl [65536,48] ----
    wmma_gemm_kernel<3, DINNER, 0, DBLP><<<dim3(M1 / 128, 1), 256, 0, stream>>>(
        u_h, xpw_h, dbl, nullptr, nullptr);

    // ---- 5) delta = softplus(dt @ dt_w^T + dt_b) ----
    delta_kernel<<<(size_t)M1 * DINNER / 256, 256, 0, stream>>>(dbl, dt_w, dt_b, delta);

    // ---- 6) chunked selective scan ----
    scan_phase1<<<NBATCH * NCHUNK, 128, 0, stream>>>(dbl, delta, u_f, A_log, Pc, Hc);
    scan_phase2<<<(NBATCH * DINNER * DSTATE) / 256, 256, 0, stream>>>(Pc, Hc, Hstart);
    scan_phase3<<<NBATCH * NCHUNK, 128, 0, stream>>>(dbl, delta, u_f, xz, A_log, Dp,
                                                     Hstart, ycore_h);

    // ---- 7) out GEMM + residual skip: ym = ycore @ out_w^T + skip*stacked ----
    wmma_gemm_kernel<4, DINNER, 1, DMODEL><<<dim3(M1 / 128, 1), 256, 0, stream>>>(
        ycore_h, ow_h, ym, skip_s, stacked_f);

    // ---- 8) regroup branches + LayerNorm ----
    regroup_ln_kernel<<<MBL, 256, 0, stream>>>(ym, norm_g, norm_b, xm_h);

    // ---- 9) final proj GEMM + bias, NCHW transpose store ----
    wmma_gemm_kernel<4, C_CH, 2, C_CH><<<dim3(MBL / 128, 256 / 64), 256, 0, stream>>>(
        xm_h, pw_h, out, proj_b, nullptr);
}
